// Distance_61667140436151
// MI455X (gfx1250) — matrix operand order repaired and checked by reading, not compiled, not run on hardware
//
#include <hip/hip_runtime.h>
#include <hip/hip_bf16.h>
#include <math.h>

typedef __attribute__((ext_vector_type(8)))  __bf16 v8bf;
typedef __attribute__((ext_vector_type(16))) __bf16 v16bf;
typedef __attribute__((ext_vector_type(8)))  float  v8f;

namespace {
constexpr int kB = 4;
constexpr int kN = 4096;   // rows of inputs1
constexpr int kM = 4096;   // rows of inputs2
constexpr int kD = 256;    // feature dim (K of the GEMM)
constexpr int BM = 128;    // output-row tile (inputs1 rows per block)
constexpr int BN = 128;    // output-col tile (inputs2 rows per block)
constexpr int BK = 32;     // K per WMMA step
constexpr int LDB = 40;    // LDS row stride in bf16 elems (32 + 8 pad, conflict-free)
constexpr float kClipMin = 1e-12f;
constexpr float kClipMax = 1e12f;
constexpr size_t kElems = (size_t)kB * kN * kD;   // per tensor
}

// Pack the high 16 bits (bf16 truncation) of two fp32 bit patterns into one
// dword: result = { hi16(u1) , hi16(u0) }  (u0 -> low half). One v_perm_b32.
__device__ __forceinline__ unsigned packhi2(unsigned u0, unsigned u1) {
#if __has_builtin(__builtin_amdgcn_perm)
  return __builtin_amdgcn_perm(u1, u0, 0x07060302u);
#else
  return (u0 >> 16) | (u1 & 0xffff0000u);
#endif
}

// Split-convert 4 fp32 -> packed bf16 hi plane (uint2) + lo plane (uint2).
// hi = truncate(x); lo = truncate(x - hi).
__device__ __forceinline__ void split4(float4 v, uint2& hi, uint2& lo) {
  unsigned ux = __float_as_uint(v.x);
  unsigned uy = __float_as_uint(v.y);
  unsigned uz = __float_as_uint(v.z);
  unsigned uw = __float_as_uint(v.w);
  hi.x = packhi2(ux, uy);
  hi.y = packhi2(uz, uw);
  float rx = v.x - __uint_as_float(ux & 0xffff0000u);
  float ry = v.y - __uint_as_float(uy & 0xffff0000u);
  float rz = v.z - __uint_as_float(uz & 0xffff0000u);
  float rw = v.w - __uint_as_float(uw & 0xffff0000u);
  lo.x = packhi2(__float_as_uint(rx), __float_as_uint(ry));
  lo.y = packhi2(__float_as_uint(rz), __float_as_uint(rw));
}

// CDNA5 async global->LDS copy, 16 bytes per lane, ASYNCcnt-tracked.
__device__ __forceinline__ void async_copy_b128(unsigned lds_off,
                                                const void* gptr) {
  asm volatile("global_load_async_to_lds_b128 %0, %1, off"
               :: "v"(lds_off), "v"((unsigned long long)(uintptr_t)gptr)
               : "memory");
}
__device__ __forceinline__ void wait_async0() {
#if __has_builtin(__builtin_amdgcn_s_wait_asynccnt)
  __builtin_amdgcn_s_wait_asynccnt(0);
#else
  asm volatile("s_wait_asynccnt 0" ::: "memory");
#endif
}

// ---------------------------------------------------------------------------
// Pre-pass: fp32 -> packed bf16 hi/lo planes for both inputs (done ONCE).
// ---------------------------------------------------------------------------
__global__ __launch_bounds__(256) void bfsplit_kernel(
    const float* __restrict__ inA, const float* __restrict__ inB,
    __bf16* __restrict__ hiA, __bf16* __restrict__ loA,
    __bf16* __restrict__ hiB, __bf16* __restrict__ loB) {
  size_t i = ((size_t)blockIdx.x * blockDim.x + threadIdx.x) * 4;   // 4 elems
  float4 va = *(const float4*)(inA + i);
  uint2 h, l;
  split4(va, h, l);
  *(uint2*)(hiA + i) = h;
  *(uint2*)(loA + i) = l;
  float4 vb = *(const float4*)(inB + i);
  split4(vb, h, l);
  *(uint2*)(hiB + i) = h;
  *(uint2*)(loB + i) = l;
}

// ---------------------------------------------------------------------------
// Main kernel, async-staging variant: LDS tiles filled with
// global_load_async_to_lds_b128 from the precomputed bf16 planes.
// ---------------------------------------------------------------------------
__global__ __launch_bounds__(256) void dist_wmma_async_kernel(
    const __bf16* __restrict__ hiA, const __bf16* __restrict__ loA,
    const __bf16* __restrict__ hiB, const __bf16* __restrict__ loB,
    const float* __restrict__ inA, const float* __restrict__ inB,
    float* __restrict__ out) {
  __shared__ __align__(16) __bf16 lds_ah[BM * LDB];
  __shared__ __align__(16) __bf16 lds_al[BM * LDB];
  __shared__ __align__(16) __bf16 lds_bh[BN * LDB];
  __shared__ __align__(16) __bf16 lds_bl[BN * LDB];
  __shared__ float lds_na[BM];
  __shared__ float lds_nb[BN];

  const int tid   = threadIdx.x;
  const int lane  = tid & 31;
  const int wave  = tid >> 5;
  const int wr    = wave >> 2;          // 0..1 : wave row group (N dim)
  const int wc    = wave & 3;           // 0..3 : wave col group (M dim)
  const int l16   = lane & 15;
  const int lhalf = (lane >> 4) & 1;

  const int b  = blockIdx.z;
  const int n0 = blockIdx.y * BM;
  const int m0 = blockIdx.x * BN;

  const float* Ab = inA + (size_t)b * kN * kD;
  const float* Bb = inB + (size_t)b * kM * kD;

  // ---- squared row norms (fp32 inputs, computed once per block) ----
  {
    const float* src = (tid < BM) ? (Ab + (size_t)(n0 + tid) * kD)
                                  : (Bb + (size_t)(m0 + (tid - BM)) * kD);
    float s = 0.f;
    #pragma unroll 4
    for (int k = 0; k < kD; k += 4) {
      float4 v = *(const float4*)(src + k);
      s = fmaf(v.x, v.x, s);
      s = fmaf(v.y, v.y, s);
      s = fmaf(v.z, v.z, s);
      s = fmaf(v.w, v.w, s);
    }
    if (tid < BM) lds_na[tid] = s;
    else          lds_nb[tid - BM] = s;
  }

  // Staging roles: wave pairs own one plane each.
  //   waves 0-1 -> A hi, 2-3 -> A lo, 4-5 -> B hi, 6-7 -> B lo.
  // Each wave issues 8 b128-async copies per K-step (4 lanes x 16B = one
  // 64-byte tile row; 8 rows per instruction; 16 instructions per plane).
  const __bf16* plane_src =
      (wave < 4) ? ((wave < 2) ? hiA : loA) : ((wave < 6) ? hiB : loB);
  const unsigned plane_dst =
      (wave < 4) ? ((wave < 2) ? (unsigned)(uintptr_t)&lds_ah[0]
                               : (unsigned)(uintptr_t)&lds_al[0])
                 : ((wave < 6) ? (unsigned)(uintptr_t)&lds_bh[0]
                               : (unsigned)(uintptr_t)&lds_bl[0]);
  const int row0g = (wave < 4) ? n0 : m0;
  const __bf16* plane_b = plane_src + (size_t)b * kN * kD;   // kN == kM
  const int r8  = lane >> 2;           // 0..7 : row within 8-row group
  const int c16 = (lane & 3) * 16;     // byte offset within 64B row

  v8f acc[4][2] = {};

  for (int k0 = 0; k0 < kD; k0 += BK) {
    __syncthreads();   // previous LDS reads done (+ norm writes visible)

    #pragma unroll
    for (int j = 0; j < 8; ++j) {
      int sub = (wave & 1) * 8 + j;          // 0..15 : 8-row group
      int row = sub * 8 + r8;                // 0..127
      unsigned ldso = plane_dst + (unsigned)(row * (LDB * 2)) + c16;
      const char* g = (const char*)plane_b +
                      ((size_t)(row0g + row) * kD + k0) * 2 + c16;
      async_copy_b128(ldso, g);
    }
    wait_async0();
    __syncthreads();

    // ---- A fragments (ISA 16-bit A 16x32 layout) ----
    const int abase = lhalf * 8;
    v16bf ah[4], al[4];
    #pragma unroll
    for (int mt = 0; mt < 4; ++mt) {
      const int arow = wr * 64 + mt * 16 + l16;
      const __bf16* ph = &lds_ah[arow * LDB];
      const __bf16* pl = &lds_al[arow * LDB];
      v8bf h0 = *(const v8bf*)(ph + abase);
      v8bf h1 = *(const v8bf*)(ph + abase + 16);
      v8bf s0 = *(const v8bf*)(pl + abase);
      v8bf s1 = *(const v8bf*)(pl + abase + 16);
      ah[mt] = __builtin_shufflevector(h0, h1, 0,1,2,3,4,5,6,7,8,9,10,11,12,13,14,15);
      al[mt] = __builtin_shufflevector(s0, s1, 0,1,2,3,4,5,6,7,8,9,10,11,12,13,14,15);
    }

    // ---- B fragments (ISA 16-bit B 32x16 layout) ----
    #pragma unroll
    for (int ct = 0; ct < 2; ++ct) {
      const int brow = wc * 32 + ct * 16 + l16;
      const __bf16* ph = &lds_bh[brow * LDB + lhalf * 8];
      const __bf16* pl = &lds_bl[brow * LDB + lhalf * 8];
      v8bf h0 = *(const v8bf*)(ph);
      v8bf h1 = *(const v8bf*)(ph + 16);
      v8bf s0 = *(const v8bf*)(pl);
      v8bf s1 = *(const v8bf*)(pl + 16);
      v16bf bh = __builtin_shufflevector(h0, h1, 0,1,2,3,4,5,6,7,8,9,10,11,12,13,14,15);
      v16bf bl = __builtin_shufflevector(s0, s1, 0,1,2,3,4,5,6,7,8,9,10,11,12,13,14,15);

      #pragma unroll
      for (int mt = 0; mt < 4; ++mt) {
        acc[mt][ct] = __builtin_amdgcn_wmma_f32_16x16x32_bf16(
            false, ah[mt], false, bh, (short)0, acc[mt][ct], false, false);
        acc[mt][ct] = __builtin_amdgcn_wmma_f32_16x16x32_bf16(
            false, ah[mt], false, bl, (short)0, acc[mt][ct], false, false);
        acc[mt][ct] = __builtin_amdgcn_wmma_f32_16x16x32_bf16(
            false, al[mt], false, bh, (short)0, acc[mt][ct], false, false);
      }
    }
  }

  // ---- epilogue ----
  float* outb = out + (size_t)b * kN * kM;
  #pragma unroll
  for (int mt = 0; mt < 4; ++mt) {
    #pragma unroll
    for (int ct = 0; ct < 2; ++ct) {
      #pragma unroll
      for (int r = 0; r < 8; ++r) {
        int rl = wr * 64 + mt * 16 + r + lhalf * 8;
        int cl = wc * 32 + ct * 16 + l16;
        float inner = acc[mt][ct][r];
        float dsq = lds_na[rl] - 2.0f * inner + lds_nb[cl];
        dsq = fminf(fmaxf(dsq, kClipMin), kClipMax);
        float val = __expf(-sqrtf(dsq));
        __builtin_nontemporal_store(
            val, &outb[(size_t)(n0 + rl) * kM + (size_t)(m0 + cl)]);
      }
    }
  }
}

// ---------------------------------------------------------------------------
// Fallback (in-loop conversion) for when workspace is too small.
// ---------------------------------------------------------------------------
__global__ __launch_bounds__(256) void dist_wmma_kernel(
    const float* __restrict__ inA, const float* __restrict__ inB,
    float* __restrict__ out) {
  __shared__ __align__(16) __bf16 lds_ah[BM * LDB];
  __shared__ __align__(16) __bf16 lds_al[BM * LDB];
  __shared__ __align__(16) __bf16 lds_bh[BN * LDB];
  __shared__ __align__(16) __bf16 lds_bl[BN * LDB];
  __shared__ float lds_na[BM];
  __shared__ float lds_nb[BN];

  const int tid   = threadIdx.x;
  const int lane  = tid & 31;
  const int wave  = tid >> 5;
  const int wr    = wave >> 2;
  const int wc    = wave & 3;
  const int l16   = lane & 15;
  const int lhalf = (lane >> 4) & 1;

  const int b  = blockIdx.z;
  const int n0 = blockIdx.y * BM;
  const int m0 = blockIdx.x * BN;

  const float* Ab = inA + (size_t)b * kN * kD;
  const float* Bb = inB + (size_t)b * kM * kD;

  {
    const float* src = (tid < BM) ? (Ab + (size_t)(n0 + tid) * kD)
                                  : (Bb + (size_t)(m0 + (tid - BM)) * kD);
    float s = 0.f;
    #pragma unroll 4
    for (int k = 0; k < kD; k += 4) {
      float4 v = *(const float4*)(src + k);
      s = fmaf(v.x, v.x, s);
      s = fmaf(v.y, v.y, s);
      s = fmaf(v.z, v.z, s);
      s = fmaf(v.w, v.w, s);
    }
    if (tid < BM) lds_na[tid] = s;
    else          lds_nb[tid - BM] = s;
  }

  v8f acc[4][2] = {};

  for (int k0 = 0; k0 < kD; k0 += BK) {
    __syncthreads();
    #pragma unroll
    for (int i = 0; i < 4; ++i) {
      int q   = tid + i * 256;
      int row = q >> 3;
      int c4  = (q & 7) << 2;
      float4 va = *(const float4*)(Ab + (size_t)(n0 + row) * kD + k0 + c4);
      uint2 ahv, alv;
      split4(va, ahv, alv);
      *(uint2*)(&lds_ah[row * LDB + c4]) = ahv;
      *(uint2*)(&lds_al[row * LDB + c4]) = alv;
      float4 vb = *(const float4*)(Bb + (size_t)(m0 + row) * kD + k0 + c4);
      uint2 bhv, blv;
      split4(vb, bhv, blv);
      *(uint2*)(&lds_bh[row * LDB + c4]) = bhv;
      *(uint2*)(&lds_bl[row * LDB + c4]) = blv;
    }
    __syncthreads();

    const int abase = lhalf * 8;
    v16bf ah[4], al[4];
    #pragma unroll
    for (int mt = 0; mt < 4; ++mt) {
      const int arow = wr * 64 + mt * 16 + l16;
      const __bf16* ph = &lds_ah[arow * LDB];
      const __bf16* pl = &lds_al[arow * LDB];
      v8bf h0 = *(const v8bf*)(ph + abase);
      v8bf h1 = *(const v8bf*)(ph + abase + 16);
      v8bf s0 = *(const v8bf*)(pl + abase);
      v8bf s1 = *(const v8bf*)(pl + abase + 16);
      ah[mt] = __builtin_shufflevector(h0, h1, 0,1,2,3,4,5,6,7,8,9,10,11,12,13,14,15);
      al[mt] = __builtin_shufflevector(s0, s1, 0,1,2,3,4,5,6,7,8,9,10,11,12,13,14,15);
    }
    #pragma unroll
    for (int ct = 0; ct < 2; ++ct) {
      const int brow = wc * 32 + ct * 16 + l16;
      const __bf16* ph = &lds_bh[brow * LDB + lhalf * 8];
      const __bf16* pl = &lds_bl[brow * LDB + lhalf * 8];
      v8bf h0 = *(const v8bf*)(ph);
      v8bf h1 = *(const v8bf*)(ph + 16);
      v8bf s0 = *(const v8bf*)(pl);
      v8bf s1 = *(const v8bf*)(pl + 16);
      v16bf bh = __builtin_shufflevector(h0, h1, 0,1,2,3,4,5,6,7,8,9,10,11,12,13,14,15);
      v16bf bl = __builtin_shufflevector(s0, s1, 0,1,2,3,4,5,6,7,8,9,10,11,12,13,14,15);
      #pragma unroll
      for (int mt = 0; mt < 4; ++mt) {
        acc[mt][ct] = __builtin_amdgcn_wmma_f32_16x16x32_bf16(
            false, ah[mt], false, bh, (short)0, acc[mt][ct], false, false);
        acc[mt][ct] = __builtin_amdgcn_wmma_f32_16x16x32_bf16(
            false, ah[mt], false, bl, (short)0, acc[mt][ct], false, false);
        acc[mt][ct] = __builtin_amdgcn_wmma_f32_16x16x32_bf16(
            false, al[mt], false, bh, (short)0, acc[mt][ct], false, false);
      }
    }
  }

  float* outb = out + (size_t)b * kN * kM;
  #pragma unroll
  for (int mt = 0; mt < 4; ++mt) {
    #pragma unroll
    for (int ct = 0; ct < 2; ++ct) {
      #pragma unroll
      for (int r = 0; r < 8; ++r) {
        int rl = wr * 64 + mt * 16 + r + lhalf * 8;
        int cl = wc * 32 + ct * 16 + l16;
        float inner = acc[mt][ct][r];
        float dsq = lds_na[rl] - 2.0f * inner + lds_nb[cl];
        dsq = fminf(fmaxf(dsq, kClipMin), kClipMax);
        float val = __expf(-sqrtf(dsq));
        __builtin_nontemporal_store(
            val, &outb[(size_t)(n0 + rl) * kM + (size_t)(m0 + cl)]);
      }
    }
  }
}

extern "C" void kernel_launch(void* const* d_in, const int* in_sizes, int n_in,
                              void* d_out, int out_size, void* d_ws, size_t ws_size,
                              hipStream_t stream) {
  const float* in1 = (const float*)d_in[0];   // inputs1 [4,4096,256] f32
  const float* in2 = (const float*)d_in[1];   // inputs2 [4,4096,256] f32
  float* out = (float*)d_out;                 // [4,4096,4096] f32

  dim3 grid(kM / BN, kN / BM, kB);            // (32, 32, 4)
  dim3 block(256);

  const size_t plane_bytes = kElems * 2;      // 8 MiB per bf16 plane
  if (ws_size >= 4 * plane_bytes) {
    __bf16* hiA = (__bf16*)d_ws;
    __bf16* loA = hiA + kElems;
    __bf16* hiB = loA + kElems;
    __bf16* loB = hiB + kElems;
    // pre-pass: 4M elems per tensor, 4 per thread
    dim3 pgrid((unsigned)(kElems / 4 / 256));
    hipLaunchKernelGGL(bfsplit_kernel, pgrid, block, 0, stream,
                       in1, in2, hiA, loA, hiB, loB);
    hipLaunchKernelGGL(dist_wmma_async_kernel, grid, block, 0, stream,
                       hiA, loA, hiB, loB, in1, in2, out);
  } else {
    hipLaunchKernelGGL(dist_wmma_kernel, grid, block, 0, stream, in1, in2, out);
  }
}
